// AffineCouplingLayer_33363305955650
// MI455X (gfx1250) — matrix operand order repaired and checked by reading, not compiled, not run on hardware
//
#include <hip/hip_runtime.h>

typedef _Float16 f16;
typedef __attribute__((ext_vector_type(16))) _Float16 v16h;
typedef __attribute__((ext_vector_type(8)))  float    v8f;

#define H 128
#define THREADS 256          // 8 waves (wave32)
#define ROWS_PER_BLOCK 128   // 8 waves * 16 rows

__global__ __launch_bounds__(THREADS)
void affine_coupling_wmma(const float* __restrict__ x,
                          const float* __restrict__ W1,
                          const float* __restrict__ b1,
                          const float* __restrict__ W2,
                          const float* __restrict__ b2,
                          const float* __restrict__ W3,
                          const float* __restrict__ b3,
                          float* __restrict__ out,
                          int Btot)
{
    // W2 transposed to [n][k] in f16 so a B-fragment is one contiguous 32B run.
    __shared__ __align__(32) f16 sW2T[H * H];       // 32 KB
    __shared__ float sW1[H], sb1[H], sb2[H];
    __shared__ float sW3[2 * H];                    // [n][j] row-major (n*2+j)
    __shared__ float sb3[2];

    const int tid = threadIdx.x;

    // ---- one-time LDS staging (uniform trip counts: no divergence) ----
    for (int idx = tid; idx < H * H; idx += THREADS) {
        const int k = idx >> 7;          // row of W2
        const int n = idx & (H - 1);     // col of W2
        sW2T[n * H + k] = (f16)W2[idx];
    }
    if (tid < H) { sW1[tid] = W1[tid]; sb1[tid] = b1[tid]; sb2[tid] = b2[tid]; }
    if (tid < 2 * H) sW3[tid] = W3[tid];
    if (tid < 2)     sb3[tid] = b3[tid];
    __syncthreads();

    const int lane  = tid & 31;
    const int wave  = tid >> 5;
    const int lhalf = lane & 15;   // 0..15
    const int upper = lane >> 4;   // 0 or 1 (lane half)

    const int numChunks = Btot / ROWS_PER_BLOCK;
    for (int chunk = blockIdx.x; chunk < numChunks; chunk += gridDim.x) {
        const int rowBase = chunk * ROWS_PER_BLOCK + wave * 16;

        // ---- layer 1 (rank-1) -> A fragments in registers ----
        // A 16x32 layout (matches B): lanes row M=lhalf, K in kb..kb+15,
        // kb = 32*kk + 16*upper.
        const int   mA  = rowBase + lhalf;
        const float zid = x[mA * 2 + 0];

        v16h aF[4];
        #pragma unroll
        for (int kk = 0; kk < 4; ++kk) {
            const int kb = kk * 32 + upper * 16;
            #pragma unroll
            for (int hi = 0; hi < 16; ++hi) {
                const int k = kb + hi;
                float v = fmaf(zid, sW1[k], sb1[k]);
                v = v > 0.f ? v : 0.f;
                aF[kk][hi] = (f16)v;
            }
        }

        // ---- layer 2: 16x128 @ 128x128 via 8 N-tiles x 4 K-steps of WMMA ----
        v8f acc[8];
        #pragma unroll
        for (int t = 0; t < 8; ++t) acc[t] = (v8f){0.f,0.f,0.f,0.f,0.f,0.f,0.f,0.f};

        #pragma unroll
        for (int t = 0; t < 8; ++t) {
            const int n = t * 16 + lhalf;     // B-fragment column this lane feeds
            #pragma unroll
            for (int kk = 0; kk < 4; ++kk) {
                // B layout: lanes 0-15 K=32kk..32kk+15, lanes 16-31 K=32kk+16..+31
                const int kb2 = kk * 32 + upper * 16;
                const v16h bF = *(const v16h*)&sW2T[n * H + kb2];
                acc[t] = __builtin_amdgcn_wmma_f32_16x16x32_f16(
                    /*neg_a=*/false, aF[kk], /*neg_b=*/false, bF,
                    /*c_mod=*/(short)0, acc[t], /*reuse_a=*/false, /*reuse_b=*/false);
            }
        }

        // ---- layer 3 fused with bias+relu: st[:,j] = relu(acc+b2) @ W3[:,j] ----
        // D layout: acc[t][v] = h2_pre[M = v + 8*upper, N = 16t + lhalf]
        float st0[8], st1[8];
        #pragma unroll
        for (int v = 0; v < 8; ++v) { st0[v] = 0.f; st1[v] = 0.f; }

        #pragma unroll
        for (int t = 0; t < 8; ++t) {
            const int n   = t * 16 + lhalf;   // same n for all 8 elems of acc[t]
            const float bias = sb2[n];
            const float w30  = sW3[n * 2 + 0];
            const float w31  = sW3[n * 2 + 1];
            #pragma unroll
            for (int v = 0; v < 8; ++v) {
                float h2 = acc[t][v] + bias;
                h2 = h2 > 0.f ? h2 : 0.f;
                st0[v] = fmaf(h2, w30, st0[v]);
                st1[v] = fmaf(h2, w31, st1[v]);
            }
        }

        // 16-lane reductions (offsets 1,2,4,8 stay within each 16-lane half)
        #pragma unroll
        for (int off = 1; off < 16; off <<= 1) {
            #pragma unroll
            for (int v = 0; v < 8; ++v) {
                st0[v] += __shfl_xor(st0[v], off, 32);
                st1[v] += __shfl_xor(st1[v], off, 32);
            }
        }

        // Select this lane's row result without dynamic register indexing.
        float r0 = 0.f, r1 = 0.f;
        #pragma unroll
        for (int v = 0; v < 8; ++v) {
            if (lhalf == v) { r0 = st0[v]; r1 = st1[v]; }
        }

        // lanes 0-7 write rows rowBase+0..7, lanes 16-23 write rows rowBase+8..15
        if (lhalf < 8) {
            const int m = rowBase + upper * 8 + lhalf;
            const float s  = tanhf(r0 + sb3[0]);
            const float tt = r1 + sb3[1];
            const float zi = x[m * 2 + 0];
            const float zt = x[m * 2 + 1];
            float2 o;
            o.x = zi;
            o.y = fmaf(zt, __expf(s), tt);
            *(float2*)&out[m * 2] = o;
        }
    }
}

extern "C" void kernel_launch(void* const* d_in, const int* in_sizes, int n_in,
                              void* d_out, int out_size, void* d_ws, size_t ws_size,
                              hipStream_t stream) {
    const float* x  = (const float*)d_in[0];
    const float* W1 = (const float*)d_in[1];
    const float* b1 = (const float*)d_in[2];
    const float* W2 = (const float*)d_in[3];
    const float* b2 = (const float*)d_in[4];
    const float* W3 = (const float*)d_in[5];
    const float* b3 = (const float*)d_in[6];
    float* out = (float*)d_out;

    const int Btot = in_sizes[0] / 2;                 // 1,048,576
    const int numChunks = Btot / ROWS_PER_BLOCK;      // 8192
    int grid = numChunks < 2048 ? numChunks : 2048;   // grid-stride, amortize W2 staging
    if (grid < 1) grid = 1;

    affine_coupling_wmma<<<grid, THREADS, 0, stream>>>(x, W1, b1, W2, b2, W3, b3, out, Btot);
}
